// VoxelSetAbstraction_44074954391637
// MI455X (gfx1250) — matrix operand order repaired and checked by reading, not compiled, not run on hardware
//
#include <hip/hip_runtime.h>
#include <hip/hip_bf16.h>

// ---------------- problem constants ----------------
#define B_      8
#define N_      16384
#define CIN_    6
#define S_      216          // 6*6*6
#define K_      32           // NSAMPLE
#define P_      (B_ * S_ * K_)   // 55296 positions for the point-phase MLP
#define Q_      (B_ * S_)         // 1728 positions for the hidden phase
#define EPS_    1e-5f

typedef _Float16 half_t;
typedef __attribute__((ext_vector_type(16))) _Float16 v16h;
typedef __attribute__((ext_vector_type(8)))  _Float16 v8h;
typedef __attribute__((ext_vector_type(8)))  float    v8f;

// ---------------------------------------------------------------------------
// prep: new_xyz (B,S,3) into ws + transposed (B,3,S) into d_out; lwh (B,3)
// lwh = [sa1, sa0, sa2] * OVERLAP / DIVIDE  (= /3 here)
// ---------------------------------------------------------------------------
__global__ void prep_kernel(const float* __restrict__ sa,
                            const float* __restrict__ pre,
                            float* __restrict__ nx,
                            float* __restrict__ lwh,
                            float* __restrict__ out_nx) {
  int t = blockIdx.x * 256 + threadIdx.x;
  if (t < B_ * 3) {
    int b = t / 3, d = t % 3;
    int pd = (d == 0) ? 1 : ((d == 1) ? 0 : 2);
    lwh[t] = sa[b * 3 + pd] * (2.0f / 6.0f);
  }
  if (t < B_ * S_ * 3) {
    int d = t % 3;
    int s = (t / 3) % S_;
    int b = t / (3 * S_);
    float v = pre[s * 3 + d] * sa[b * 3 + d] * (1.0f / 6.0f);
    nx[t] = v;
    out_nx[(size_t)b * 3 * S_ + (size_t)d * S_ + s] = v;
  }
}

// ---------------------------------------------------------------------------
// weight convert: f32 (O,Cin) -> f16 (O,Kp) zero-padded K
// ---------------------------------------------------------------------------
__global__ void convw_kernel(const float* __restrict__ W, half_t* __restrict__ dst,
                             int O, int Cin, int Kp) {
  int t = blockIdx.x * 256 + threadIdx.x;
  if (t >= O * Kp) return;
  int o = t / Kp, k = t % Kp;
  dst[t] = (k < Cin) ? (half_t)W[o * Cin + k] : (half_t)0.0f;
}

// ---------------------------------------------------------------------------
// grouping: one block per (b,s) box. Faithful to the sort-of-zeros semantics:
// Z = N - Cin zeros lead the sorted list (Cin = inside points with i>0);
// samples = min(Z,32) copies of point 0 + smallest inside indices.
// Emits X1 as f16 rows of 32 (cols 0..5 = features, 6..31 = zero K padding).
// ---------------------------------------------------------------------------
__global__ __launch_bounds__(256)
void group_kernel(const float* __restrict__ xyz,   // (B, 6, N)
                  const float* __restrict__ nx,    // (B, S, 3)
                  const float* __restrict__ lwh,   // (B, 3)
                  half_t* __restrict__ X1) {       // (P_, 32) row-major f16
  __shared__ int   s_cnt[256];
  __shared__ int   s_idx[K_];
  __shared__ float s_box[6];
  const int bs = blockIdx.x;
  const int b  = bs / S_;
  const int t  = threadIdx.x;
  if (t < 3) { s_box[t] = nx[bs * 3 + t]; s_box[3 + t] = lwh[b * 3 + t]; }
  __syncthreads();
  const float cx = s_box[0], cy = s_box[1], cz = s_box[2];
  const float lx = s_box[3], ly = s_box[4], lz = s_box[5];
  const float* px = xyz + (size_t)b * 6 * N_;

  int cnt = 0;
  for (int i = t; i < N_; i += 256) {
    float dx = fabsf(px[i] - cx) - lx;
    float dy = fabsf(px[N_ + i] - cy) - ly;
    float dz = fabsf(px[2 * N_ + i] - cz) - lz;
    bool inside = fmaxf(dx, fmaxf(dy, dz)) <= 0.0f;
    if (inside && i > 0) cnt++;
  }
  s_cnt[t] = cnt;
  __syncthreads();
  for (int off = 128; off > 0; off >>= 1) {
    if (t < off) s_cnt[t] += s_cnt[t + off];
    __syncthreads();
  }
  const int Cin = s_cnt[0];
  if (t < K_) s_idx[t] = 0;
  __syncthreads();
  const int need = Cin - (N_ - K_);   // >0 only when nearly all points inside
  if (need > 0 && t == 0) {
    int pos = K_ - need;
    for (int i = 1; i < N_ && pos < K_; ++i) {
      float dx = fabsf(px[i] - cx) - lx;
      float dy = fabsf(px[N_ + i] - cy) - ly;
      float dz = fabsf(px[2 * N_ + i] - cz) - lz;
      if (fmaxf(dx, fmaxf(dy, dz)) <= 0.0f) s_idx[pos++] = i;
    }
  }
  __syncthreads();
  for (int e = t; e < K_ * 32; e += 256) {
    int k = e >> 5, c = e & 31;
    int id = s_idx[k];
    float v = 0.0f;
    if (c < 3)       v = px[(size_t)c * N_ + id] - s_box[c];
    else if (c < 6)  v = px[(size_t)c * N_ + id];
    X1[(size_t)(bs * K_ + k) * 32 + c] = (half_t)v;
  }
}

// ---------------------------------------------------------------------------
// WMMA GEMM + bias + BN + ReLU.
//   X: (P, K) f16 row-major, W: (O, K) f16 row-major, Y: (P, O) f16 row-major
//   Y[p,o] = relu((sum_k W[o,k]*X[p,k] + bias[o]) * g[o]/sqrt(v[o]+eps)
//                 + beta[o] - mean[o]*g[o]/sqrt(v[o]+eps))
// Block = 128 threads = 4 waves; wave w -> M-tile 16 rows, 4 N-subtiles of 16.
// K multiple of 32 so the loop is pure v_wmma_f32_16x16x32_f16.
// ---------------------------------------------------------------------------
__global__ __launch_bounds__(128)
void gemm_bnrelu_kernel(const half_t* __restrict__ X,
                        const half_t* __restrict__ Wt,
                        const float* __restrict__ bias,
                        const float* __restrict__ g,
                        const float* __restrict__ beta,
                        const float* __restrict__ mean,
                        const float* __restrict__ var,
                        half_t* __restrict__ Y,
                        int P, int K, int O) {
  const int lane   = threadIdx.x & 31;
  const int wave   = threadIdx.x >> 5;
  const int m_tile = blockIdx.y * 64 + wave * 16;
  const int n_base = blockIdx.x * 64;
  const int h      = lane >> 4;    // K-half selector (ISA lane-group split)
  const int lm     = lane & 15;

  v8f acc[4];
#pragma unroll
  for (int ns = 0; ns < 4; ++ns)
#pragma unroll
    for (int r = 0; r < 8; ++r) acc[ns][r] = 0.0f;

  for (int kk = 0; kk < K; kk += 32) {
    // A fragment (W): lane = row m_tile+lm; element e -> K = h*16 + e (contiguous)
    const half_t* arow = Wt + (size_t)(m_tile + lm) * K + kk + h * 16;
    v8h alo = *(const v8h*)(arow);
    v8h ahi = *(const v8h*)(arow + 8);
    v16h a;
#pragma unroll
    for (int e = 0; e < 8; ++e) { a[e] = alo[e]; a[e + 8] = ahi[e]; }

#pragma unroll
    for (int ns = 0; ns < 4; ++ns) {
      // B fragment (X): lane = column n; element e -> K = h*16 + e (contiguous)
      const int n = n_base + ns * 16 + lm;
      v16h bfr = *(const v16h*)(X + (size_t)n * K + kk + h * 16);
      acc[ns] = __builtin_amdgcn_wmma_f32_16x16x32_f16(
          false, a, false, bfr, (short)0, acc[ns], false, false);
    }
  }

  // fused bias + inference-BN + ReLU epilogue
  float sc[8], sh[8], bs[8];
#pragma unroll
  for (int r = 0; r < 8; ++r) {
    const int o = m_tile + r + 8 * h;
    const float s = g[o] * rsqrtf(var[o] + EPS_);
    sc[r] = s;
    sh[r] = beta[o] - mean[o] * s;
    bs[r] = bias[o];
  }
#pragma unroll
  for (int ns = 0; ns < 4; ++ns) {
    const int p = n_base + ns * 16 + lm;
#pragma unroll
    for (int r = 0; r < 8; ++r) {
      const int o = m_tile + r + 8 * h;
      float y = (acc[ns][r] + bs[r]) * sc[r] + sh[r];
      y = fmaxf(y, 0.0f);
      Y[(size_t)p * O + o] = (half_t)y;
    }
  }
}

// ---------------------------------------------------------------------------
// max over the K=32 sample axis: (Q*32, 256) -> (Q, 256)
// ---------------------------------------------------------------------------
__global__ void pool_kernel(const half_t* __restrict__ Y3, half_t* __restrict__ pooled) {
  int t = blockIdx.x * 256 + threadIdx.x;
  if (t >= Q_ * 256) return;
  int q = t >> 8, c = t & 255;
  const half_t* base = Y3 + (size_t)q * K_ * 256 + c;
  float m = (float)base[0];
  for (int k = 1; k < K_; ++k) m = fmaxf(m, (float)base[(size_t)k * 256]);
  pooled[(size_t)q * 256 + c] = (half_t)m;
}

// ---------------------------------------------------------------------------
// finalize: (Q, 512) f16 -> d_out[(B,512,S)] f32 (after the 5184-elem new_xyz)
// ---------------------------------------------------------------------------
__global__ void finalize_kernel(const half_t* __restrict__ H2, float* __restrict__ out) {
  int t = blockIdx.x * 256 + threadIdx.x;
  if (t >= Q_ * 512) return;
  int o = t & 511, q = t >> 9;
  int b = q / S_, s = q % S_;
  out[B_ * 3 * S_ + (size_t)b * 512 * S_ + (size_t)o * S_ + s] =
      (float)H2[(size_t)q * 512 + o];
}

// ---------------------------------------------------------------------------
extern "C" void kernel_launch(void* const* d_in, const int* in_sizes, int n_in,
                              void* d_out, int out_size, void* d_ws, size_t ws_size,
                              hipStream_t stream) {
  (void)in_sizes; (void)n_in; (void)out_size; (void)ws_size;

  const float* xyz = (const float*)d_in[0];
  const float* sa  = (const float*)d_in[1];
  const float* pre = (const float*)d_in[2];
  // mlp_params: 3 layers x (W,b,g,beta,m,v) at d_in[3..20]; hidden at [21..32]
  const float* Wp[5]; const float* bp[5]; const float* gp[5];
  const float* be[5]; const float* mp[5]; const float* vp[5];
  for (int l = 0; l < 5; ++l) {
    Wp[l] = (const float*)d_in[3 + 6 * l + 0];
    bp[l] = (const float*)d_in[3 + 6 * l + 1];
    gp[l] = (const float*)d_in[3 + 6 * l + 2];
    be[l] = (const float*)d_in[3 + 6 * l + 3];
    mp[l] = (const float*)d_in[3 + 6 * l + 4];
    vp[l] = (const float*)d_in[3 + 6 * l + 5];
  }
  float* out = (float*)d_out;
  char* ws = (char*)d_ws;

  // workspace layout (all on-chip-resident: ~45 MB << 192 MB L2)
  float*  nx   = (float*)(ws + 0);          // 5184 f32
  float*  lwh  = (float*)(ws + 20992);      // 24 f32
  half_t* wbuf = (half_t*)(ws + 21248);     // 436224 f16 padded weights
  half_t* bufA = (half_t*)(ws + (1u << 20));            // <= 14.2 MB
  half_t* bufB = (half_t*)(ws + (16u << 20));           // <= 28.4 MB

  // padded f16 weight sub-buffers (halfs)
  const int Kp[5] = {32, 64, 128, 256, 512};
  const int Oo[5] = {64, 128, 256, 512, 512};
  const int Ci[5] = {CIN_, 64, 128, 256, 512};
  size_t woff[5]; size_t acc = 0;
  for (int l = 0; l < 5; ++l) { woff[l] = acc; acc += (size_t)Oo[l] * Kp[l]; }

  // 1) prep
  prep_kernel<<<(B_ * S_ * 3 + 255) / 256, 256, 0, stream>>>(sa, pre, nx, lwh, out);

  // 2) weight conversion (f32 -> zero-K-padded f16)
  for (int l = 0; l < 5; ++l) {
    int n = Oo[l] * Kp[l];
    convw_kernel<<<(n + 255) / 256, 256, 0, stream>>>(Wp[l], wbuf + woff[l],
                                                      Oo[l], Ci[l], Kp[l]);
  }

  // 3) grouping -> X1 (P_, 32) f16 in bufA
  group_kernel<<<Q_, 256, 0, stream>>>(xyz, nx, lwh, bufA);

  // 4) point-phase MLP: ping-pong bufA/bufB, all WMMA
  // L1: (P_,32)x(64,32) -> bufB (P_,64)
  gemm_bnrelu_kernel<<<dim3(P_ / 64, Oo[0] / 64), 128, 0, stream>>>(
      bufA, wbuf + woff[0], bp[0], gp[0], be[0], mp[0], vp[0], bufB, P_, Kp[0], Oo[0]);
  // L2: (P_,64)x(128,64) -> bufA (P_,128)
  gemm_bnrelu_kernel<<<dim3(P_ / 64, Oo[1] / 64), 128, 0, stream>>>(
      bufB, wbuf + woff[1], bp[1], gp[1], be[1], mp[1], vp[1], bufA, P_, Kp[1], Oo[1]);
  // L3: (P_,128)x(256,128) -> bufB (P_,256)
  gemm_bnrelu_kernel<<<dim3(P_ / 64, Oo[2] / 64), 128, 0, stream>>>(
      bufA, wbuf + woff[2], bp[2], gp[2], be[2], mp[2], vp[2], bufB, P_, Kp[2], Oo[2]);

  // 5) max over samples -> pooled (Q_,256) in bufA
  pool_kernel<<<(Q_ * 256 + 255) / 256, 256, 0, stream>>>(bufB, bufA);

  // 6) hidden FCs
  // H1: (Q_,256)x(512,256) -> bufB (Q_,512)
  gemm_bnrelu_kernel<<<dim3(Q_ / 64, Oo[3] / 64), 128, 0, stream>>>(
      bufA, wbuf + woff[3], bp[3], gp[3], be[3], mp[3], vp[3], bufB, Q_, Kp[3], Oo[3]);
  // H2: (Q_,512)x(512,512) -> bufA (Q_,512)
  gemm_bnrelu_kernel<<<dim3(Q_ / 64, Oo[4] / 64), 128, 0, stream>>>(
      bufB, wbuf + woff[4], bp[4], gp[4], be[4], mp[4], vp[4], bufA, Q_, Kp[4], Oo[4]);

  // 7) transpose/widen to f32 output (B,512,S)
  finalize_kernel<<<(Q_ * 512 + 255) / 256, 256, 0, stream>>>(bufA, out);
}
